// PrunedMoELayer_77936476553721
// MI455X (gfx1250) — compile-verified, matrix operand order checked by reading, not run on hardware
//
#include <hip/hip_runtime.h>

typedef __attribute__((ext_vector_type(16))) __bf16 v16bf;
typedef __attribute__((ext_vector_type(2)))  __bf16 v2bf;
typedef __attribute__((ext_vector_type(8)))  float  v8f;
typedef __attribute__((ext_vector_type(2)))  float  v2f;
typedef int v4i __attribute__((vector_size(16)));
typedef __attribute__((address_space(3))) v4i lds_v4i;
typedef __attribute__((address_space(1))) v4i glb_v4i;

constexpr int Tt = 8192;   // B*N tokens
constexpr int Dd = 1024;   // model dim
constexpr int Hh = 2048;   // hidden dim (2D)
constexpr int Ee = 8;      // experts

constexpr int BM = 256, BN = 128, BK = 32;
constexpr int SAS = BK + 8;      // padded LDS row stride: 40 elems = 80 B (16B aligned)
constexpr int THREADS = 512;     // 16 waves: 8 (M) x 2 (N)

// packed fp32x2 -> bf16x2 (single v_cvt_pk_bf16_f32)
__device__ inline unsigned pk2(float a, float b) {
  v2f f = {a, b};
  union { v2bf h; unsigned u; } u;
  u.h = __builtin_convertvector(f, v2bf);
  return u.u;
}

// 16-byte async copy global -> LDS (ASYNCcnt-tracked, bypasses VGPRs)
__device__ inline void async_ld16(const void* gptr, void* lptr) {
  lds_v4i* lp = (lds_v4i*)lptr;  // addrspacecast generic -> LDS
  glb_v4i* gp = (glb_v4i*)(unsigned long long)(__SIZE_TYPE__)gptr;
#if __has_builtin(__builtin_amdgcn_global_load_async_to_lds_b128)
  __builtin_amdgcn_global_load_async_to_lds_b128(gp, lp, 0, 0);
#else
  asm volatile("global_load_async_to_lds_b128 %0, %1, off"
               :: "v"(lp), "v"(gp) : "memory");
#endif
}

__device__ inline void wait_async0() {
#if __has_builtin(__builtin_amdgcn_s_wait_asynccnt)
  __builtin_amdgcn_s_wait_asynccnt(0);
#else
  asm volatile("s_wait_asynccnt 0x0" ::: "memory");
#endif
}

// ---------------- gating: argmax(logits + gumbel) ----------------
__global__ __launch_bounds__(256)
void gate_kernel(const float* __restrict__ x, const float* __restrict__ Wg,
                 const float* __restrict__ bg, const float* __restrict__ gum,
                 int* __restrict__ expert, int* __restrict__ counts) {
  const int wave = threadIdx.x >> 5;
  const int lane = threadIdx.x & 31;
  const int t = blockIdx.x * 8 + wave;   // one wave32 per token
  float acc[Ee];
#pragma unroll
  for (int e = 0; e < Ee; ++e) acc[e] = 0.f;
  const float* xr = x + (size_t)t * Dd;
  for (int k = lane; k < Dd; k += 32) {
    const float xv = xr[k];
    const float* wr = Wg + (size_t)k * Ee;
#pragma unroll
    for (int e = 0; e < Ee; ++e) acc[e] += xv * wr[e];
  }
#pragma unroll
  for (int off = 16; off > 0; off >>= 1) {
#pragma unroll
    for (int e = 0; e < Ee; ++e) acc[e] += __shfl_down(acc[e], off, 32);
  }
  if (lane == 0) {
    int best = 0; float bestv = -3.4e38f;
#pragma unroll
    for (int e = 0; e < Ee; ++e) {
      const float v = acc[e] + bg[e] + gum[(size_t)t * Ee + e];
      if (v > bestv) { bestv = v; best = e; }
    }
    expert[t] = best;
    atomicAdd(&counts[best], 1);
  }
}

__global__ void scan_kernel(const int* __restrict__ counts, int* __restrict__ offsets) {
  if (threadIdx.x == 0) {
    int s = 0;
    for (int e = 0; e < Ee; ++e) { offsets[e] = s; s += counts[e]; }
    offsets[Ee] = s;
  }
}

__global__ __launch_bounds__(256)
void scatter_kernel(const int* __restrict__ expert, const int* __restrict__ offsets,
                    int* __restrict__ cursors, int* __restrict__ perm) {
  const int t = blockIdx.x * blockDim.x + threadIdx.x;
  if (t >= Tt) return;
  const int e = expert[t];
  const int p = atomicAdd(&cursors[e], 1);
  perm[offsets[e] + p] = t;
}

// ------- grouped GEMM, double-buffered + pipelined, WMMA bf16 256x128x32 -------
// FIRST:  h = relu(gather(x, perm_seg) @ W1[e] + b1[e])   (fp32 in, bf16 out)
// !FIRST: out[perm] = h_seg @ W2[e] + b2[e]  (bf16 in via async-to-LDS, fp32 out)
template <bool FIRST>
__global__ __launch_bounds__(THREADS)
void moe_gemm(const void* __restrict__ Abase,
              const float* __restrict__ W,
              const float* __restrict__ bias,
              const int* __restrict__ perm,
              const int* __restrict__ offsets,
              unsigned short* __restrict__ h_out,
              float* __restrict__ out)
{
  constexpr int K  = FIRST ? Dd : Hh;
  constexpr int Nn = FIRST ? Hh : Dd;

  __shared__ __bf16 sA[2][BM * SAS];
  __shared__ __bf16 sB[2][BN * SAS];

  const int e = blockIdx.z;
  const int seg_start = offsets[e];
  const int seg_count = offsets[e + 1] - seg_start;
  const int m_base = blockIdx.y * BM;
  if (m_base >= seg_count) return;               // skip empty tiles
  const int n_base = blockIdx.x * BN;

  const float* We = W + (size_t)e * K * Nn;

  const int tid  = threadIdx.x;
  const int wave = tid >> 5, lane = tid & 31;
  const int wm = wave & 7, wn = wave >> 3;       // 8x2 waves -> 32x64 per wave
  const int lrow = lane & 15, hi = lane >> 4;
  const int kb = hi * 8;                          // per-half K sub-chunk

  v8f acc[2][4];
  const v8f vz = {0.f, 0.f, 0.f, 0.f, 0.f, 0.f, 0.f, 0.f};
#pragma unroll
  for (int i = 0; i < 2; ++i)
#pragma unroll
    for (int j = 0; j < 4; ++j) acc[i][j] = vz;

  // B staging coords: 32 x 128 fp32 tile, 8 fp32 per thread
  const int bkr = tid >> 5;          // 0..15
  const int bc4 = (tid & 31) * 4;    // 0..124
  // A staging coords
  const int ar0_f = tid >> 3;        // FIRST: 0..63 (rows +i*64)
  const int aco_f = (tid & 7) * 4;
  const int ar0_s = tid >> 2;        // !FIRST: 0..127 (rows +i*128)
  const int ac8_s = (tid & 3) * 8;

  // branchless row gather: clamp OOB rows (their C rows are discarded anyway)
  int prow[4];
  if (FIRST) {
#pragma unroll
    for (int i = 0; i < 4; ++i) {
      int rp = m_base + ar0_f + i * 64;
      rp = rp < seg_count ? rp : seg_count - 1;
      prow[i] = perm[seg_start + rp];
    }
  }
  int arow_s[2];
  if (!FIRST) {
#pragma unroll
    for (int i = 0; i < 2; ++i) {
      int rp = m_base + ar0_s + i * 128;
      rp = rp < seg_count ? rp : seg_count - 1;
      arow_s[i] = seg_start + rp;
    }
  }

  float4 areg[4], breg[2];
  auto fetchB = [&](int k0) {
#pragma unroll
    for (int i = 0; i < 2; ++i)
      breg[i] = *(const float4*)(We + (size_t)(k0 + bkr + i * 16) * Nn + n_base + bc4);
  };
  auto fetchA_f = [&](int k0) {
    const float* xf = (const float*)Abase;
#pragma unroll
    for (int i = 0; i < 4; ++i)
      areg[i] = *(const float4*)(xf + (size_t)prow[i] * Dd + k0 + aco_f);
  };
  auto issueA_s = [&](int k0, int b) {
    const unsigned short* hf = (const unsigned short*)Abase;
#pragma unroll
    for (int i = 0; i < 2; ++i)
      async_ld16(hf + (size_t)arow_s[i] * Hh + k0 + ac8_s,
                 &sA[b][(ar0_s + i * 128) * SAS + ac8_s]);
  };

  // prologue: stage tile 0
  fetchB(0);
  if (FIRST) fetchA_f(0); else issueA_s(0, 0);

  int buf = 0;
  for (int k0 = 0; k0 < K; k0 += BK) {
    // ---- commit staged tile to LDS[buf] ----
    if (FIRST) {
#pragma unroll
      for (int i = 0; i < 4; ++i) {
        uint2 p = make_uint2(pk2(areg[i].x, areg[i].y), pk2(areg[i].z, areg[i].w));
        *(uint2*)&sA[buf][(ar0_f + i * 64) * SAS + aco_f] = p;
      }
    }
#pragma unroll
    for (int i = 0; i < 2; ++i) {
      const int k = bkr + i * 16;
      sB[buf][(bc4 + 0) * SAS + k] = (__bf16)breg[i].x;
      sB[buf][(bc4 + 1) * SAS + k] = (__bf16)breg[i].y;
      sB[buf][(bc4 + 2) * SAS + k] = (__bf16)breg[i].z;
      sB[buf][(bc4 + 3) * SAS + k] = (__bf16)breg[i].w;
    }
    if (!FIRST) wait_async0();     // async A tile for this buffer must be in LDS
    __syncthreads();

    // ---- issue next tile's fetches; they overlap the WMMAs below ----
    if (k0 + BK < K) {
      fetchB(k0 + BK);
      if (FIRST) fetchA_f(k0 + BK); else issueA_s(k0 + BK, buf ^ 1);
      if (k0 + 2 * BK < K)
        __builtin_prefetch(We + (size_t)(k0 + 2 * BK + bkr) * Nn + n_base + bc4, 0, 1);
    }

    // ---- fragments: lane<16 -> K 0..7 & 16..23 ; lane>=16 -> K 8..15 & 24..31 ----
    v16bf afrag[2], bfrag[4];
#pragma unroll
    for (int i = 0; i < 2; ++i) {
      const __bf16* base = &sA[buf][(wm * 32 + i * 16 + lrow) * SAS];
      ((uint4*)&afrag[i])[0] = *(const uint4*)(base + kb);
      ((uint4*)&afrag[i])[1] = *(const uint4*)(base + 16 + kb);
    }
#pragma unroll
    for (int j = 0; j < 4; ++j) {
      const __bf16* base = &sB[buf][(wn * 64 + j * 16 + lrow) * SAS];
      ((uint4*)&bfrag[j])[0] = *(const uint4*)(base + kb);
      ((uint4*)&bfrag[j])[1] = *(const uint4*)(base + 16 + kb);
    }
#pragma unroll
    for (int i = 0; i < 2; ++i)
#pragma unroll
      for (int j = 0; j < 4; ++j)
        acc[i][j] = __builtin_amdgcn_wmma_f32_16x16x32_bf16(
            false, afrag[i], false, bfrag[j], (short)0, acc[i][j], false, false);

    buf ^= 1;
  }

  // ---- epilogue: C layout = VGPR r -> M = hi*8 + r, N = lane&15 ----
#pragma unroll
  for (int i = 0; i < 2; ++i) {
#pragma unroll
    for (int j = 0; j < 4; ++j) {
#pragma unroll
      for (int r = 0; r < 8; ++r) {
        const int row = wm * 32 + i * 16 + hi * 8 + r;
        const int col = wn * 64 + j * 16 + lrow;
        const int pos = m_base + row;
        if (pos >= seg_count) continue;
        const int gcol = n_base + col;
        float v = acc[i][j][r] + bias[(size_t)e * Nn + gcol];
        if (FIRST) {
          v = fmaxf(v, 0.f);
          __bf16 bv = (__bf16)v;
          h_out[(size_t)(seg_start + pos) * Hh + gcol] = *(unsigned short*)&bv;
        } else {
          const int tok = perm[seg_start + pos];
          out[(size_t)tok * Dd + gcol] = v;
        }
      }
    }
  }
}

extern "C" void kernel_launch(void* const* d_in, const int* in_sizes, int n_in,
                              void* d_out, int out_size, void* d_ws, size_t ws_size,
                              hipStream_t stream) {
  const float* x   = (const float*)d_in[0];
  const float* gum = (const float*)d_in[1];
  const float* Wg  = (const float*)d_in[2];
  const float* bg  = (const float*)d_in[3];
  const float* W1  = (const float*)d_in[4];
  const float* b1  = (const float*)d_in[5];
  const float* W2  = (const float*)d_in[6];
  const float* b2  = (const float*)d_in[7];
  float* out = (float*)d_out;

  char* ws = (char*)d_ws;
  int* counts  = (int*)ws;          // 8
  int* cursors = counts + 8;        // 8
  int* offsets = cursors + 8;       // 9
  int* expert  = (int*)(ws + 128);                  // Tt ints
  int* perm    = expert + Tt;                       // Tt ints
  unsigned short* h = (unsigned short*)(ws + 128 + 2 * Tt * (int)sizeof(int) + 128);
  // h: Tt * Hh bf16 = 32 MB

  (void)in_sizes; (void)n_in; (void)out_size; (void)ws_size;

  (void)hipMemsetAsync(d_ws, 0, 128, stream);      // zero counts + cursors
  gate_kernel<<<Tt / 8, 256, 0, stream>>>(x, Wg, bg, gum, expert, counts);
  scan_kernel<<<1, 32, 0, stream>>>(counts, offsets);
  scatter_kernel<<<Tt / 256, 256, 0, stream>>>(expert, offsets, cursors, perm);
  moe_gemm<true><<<dim3(Hh / BN, Tt / BM, Ee), THREADS, 0, stream>>>(
      (const void*)x, W1, b1, perm, offsets, h, nullptr);
  moe_gemm<false><<<dim3(Dd / BN, Tt / BM, Ee), THREADS, 0, stream>>>(
      (const void*)h, W2, b2, perm, offsets, nullptr, out);
}
